// DiffHistogram_12962211299681
// MI455X (gfx1250) — matrix operand order repaired and checked by reading, not compiled, not run on hardware
//
#include <hip/hip_runtime.h>

typedef __attribute__((ext_vector_type(16))) _Float16 v16h;
typedef __attribute__((ext_vector_type(8)))  float    v8f;

#define NBINS 32
#define HW 65536               // 256*256 pixels per channel
#define NCH 64                 // 8*8 channels
#define BLOCKS_PER_CH 16
#define WAVES_PER_BLOCK 8
#define PIX_PER_WAVE (HW / (BLOCKS_PER_CH * WAVES_PER_BLOCK))   // 512
#define ITERS (PIX_PER_WAVE / 16)                               // 32

__global__ void diffhist_zero_kernel(float* __restrict__ out, int n) {
    int i = blockIdx.x * blockDim.x + threadIdx.x;
    if (i < n) out[i] = 0.0f;
}

__global__ __launch_bounds__(256)
void DiffHistogram_12962211299681_kernel(const float* __restrict__ x,
                                         const float* __restrict__ centers,
                                         float* __restrict__ out) {
    const int lane = threadIdx.x & 31;
    const int wave = threadIdx.x >> 5;
    const int ch   = blockIdx.x;          // 0..63  == b*8 + c
    const int blk  = blockIdx.y;          // 0..15  within channel

    const float S       = 1.0f / 2.5066f; // ER / RATIO
    const float inv2s2  = 512.0f;         // 1/(2*sigma^2), sigma = 1/32

    const float c_lo = centers[0];
    const float c_hi = centers[NBINS - 1];

    // Bins owned by this lane per the 16-bit A-matrix (16x32) VGPR layout:
    //   lanes 0-15 : K = {0..7, 16..23};  lanes 16-31 : K = {8..15, 24..31}
    const int binBase = (lane >> 4) * 8;
    float myC[16];
#pragma unroll
    for (int e = 0; e < 16; ++e) {
        int bin = binBase + ((e < 8) ? e : (e + 8));
        myC[e] = centers[bin];
    }

    // Constant selection B matrices (f16 32x16, column n = lane&15):
    //   B0[k,n] = (k == n)       -> selects bins 0..15
    //   B1[k,n] = (k == n + 16)  -> selects bins 16..31
    // Layout: lanes 0-15 hold K={0..7,16..23}, lanes 16-31 hold K={8..15,24..31}.
    v16h b0, b1;
#pragma unroll
    for (int e = 0; e < 16; ++e) {
        int kk = binBase + ((e < 8) ? e : (e + 8));
        b0[e] = (_Float16)((kk == (lane & 15)) ? 1.0f : 0.0f);
        b1[e] = (_Float16)((kk == (lane & 15) + 16) ? 1.0f : 0.0f);
    }

    v8f c0 = {};   // accumulates per-bin sums for bins 0..15
    v8f c1 = {};   // accumulates per-bin sums for bins 16..31

    const float* xp = x + (size_t)ch * HW
                        + (size_t)blk * (HW / BLOCKS_PER_CH)
                        + (size_t)wave * PIX_PER_WAVE;
    const int m = lane & 15;   // pixel slot; lane pair (m, m+16) shares a pixel

    for (int it = 0; it < ITERS; ++it) {
        float xv = xp[it * 16 + m];
        xv = fminf(fmaxf(xv, c_lo), c_hi);

        v16h a;
#pragma unroll
        for (int e = 0; e < 16; ++e) {
            float d = xv - myC[e];
            float w = S * __expf(-(d * d) * inv2s2);
            a[e] = (_Float16)w;
        }

        // C[m,n] += w(pixel m, bin n): matrix pipe performs the running
        // reduction over iterations while VALU/TRANS computes the next tile.
        c0 = __builtin_amdgcn_wmma_f32_16x16x32_f16(false, a, false, b0,
                                                    (short)0, c0, false, false);
        c1 = __builtin_amdgcn_wmma_f32_16x16x32_f16(false, a, false, b1,
                                                    (short)0, c1, false, false);
    }

    // Reduce accumulator over the M dimension:
    //  - 8 VGPRs per lane cover M = 0..7 (lanes 0-15) / M = 8..15 (lanes 16-31)
    float s0 = 0.0f, s1 = 0.0f;
#pragma unroll
    for (int r = 0; r < 8; ++r) { s0 += c0[r]; s1 += c1[r]; }
    //  - add the other half-wave's partial (M halves)
    s0 += __shfl_xor(s0, 16, 32);
    s1 += __shfl_xor(s1, 16, 32);

    // lane n (0..15) now holds this wave's total for bin n / bin n+16
    if (lane < 16) {
        atomicAdd(&out[ch * NBINS + lane],      s0);
        atomicAdd(&out[ch * NBINS + 16 + lane], s1);
    }
}

extern "C" void kernel_launch(void* const* d_in, const int* in_sizes, int n_in,
                              void* d_out, int out_size, void* d_ws, size_t ws_size,
                              hipStream_t stream) {
    const float* x       = (const float*)d_in[0];   // [8,8,256,256] f32
    const float* centers = (const float*)d_in[1];   // [32] f32
    float* out = (float*)d_out;                     // [8,256,1,1] = 2048 f32

    // d_out is poisoned by the harness; we accumulate with atomics, so zero it.
    diffhist_zero_kernel<<<(out_size + 255) / 256, 256, 0, stream>>>(out, out_size);

    dim3 grid(NCH, BLOCKS_PER_CH);
    DiffHistogram_12962211299681_kernel<<<grid, 256, 0, stream>>>(x, centers, out);
}
